// UMABlock_27547920236677
// MI455X (gfx1250) — matrix operand order repaired and checked by reading, not compiled, not run on hardware
//
#include <hip/hip_runtime.h>

// ---------------------------------------------------------------------------
// UMA / eSCN block for MI455X (gfx1250, wave32, WMMA bf16 -> f32 accumulate)
// v3: - wigner_fwd stages the per-edge rad vector (3072 bf16) into LDS via
//       b128 loads; epilogue reads LDS instead of 8K scalar global u16 loads.
//     - recomb stores y2 coalesced [e][19][128]; wigner_inv does the
//       transpose during LDS staging (b128 row loads, b16 LDS scatter).
//     - GEMM keeps CDNA5 async global->LDS double-buffered staging
//       (GLOBAL_LOAD_ASYNC_TO_LDS_B128 / s_wait_asynccnt).
// ---------------------------------------------------------------------------

typedef __attribute__((ext_vector_type(16))) __bf16 v16bf;
typedef __attribute__((ext_vector_type(8)))  float  v8f;

#define WMMA_BF16(A,B,C) \
  __builtin_amdgcn_wmma_f32_16x16x32_bf16(false,(A),false,(B),(short)0,(C),false,false)

static __device__ __forceinline__ float  bf2f(__bf16 v){ return (float)v; }
static __device__ __forceinline__ __bf16 f2bf(float v){ return (__bf16)v; }
static __device__ __forceinline__ float  siluf(float x){ return x * (1.0f/(1.0f+__expf(-x))); }
static __device__ __forceinline__ float  sigmf(float x){ return 1.0f/(1.0f+__expf(-x)); }

// low 32 bits of a generic pointer to LDS == LDS byte offset (aperture layout)
static __device__ __forceinline__ unsigned lds_addr(const void* p){
  return (unsigned)(size_t)p;
}
// CDNA5 async copy: LDS[%0] <= MEM[%1] (16 bytes), tracked by ASYNCcnt
static __device__ __forceinline__ void async_b128(unsigned lds, const void* g){
  asm volatile("global_load_async_to_lds_b128 %0, %1, off"
               :: "v"(lds), "v"(g) : "memory");
}

// A fragment (16x32 bf16, [M][K] row-major tile in LDS, row stride = ld):
// lanes 0-15: M=lane, K=0..7 & 16..23 ; lanes 16-31: M=lane-16, K=8..15 & 24..31
static __device__ __forceinline__ v16bf frag_a(const __bf16* t, int ld, int mbase, int k0){
  int lane = threadIdx.x & 31;
  const __bf16* p = t + (mbase + (lane&15))*ld + k0 + ((lane&16)?8:0);
  v16bf v;
  ((int4*)&v)[0] = *(const int4*)(p);
  ((int4*)&v)[1] = *(const int4*)(p+16);
  return v;
}
// B fragment (32x16 bf16, pre-transposed [N][K] tile in LDS):
// lanes 0-15: N=lane, K=0..7 & 16..23 ; lanes 16-31: N=lane-16, K=8..15 & 24..31
static __device__ __forceinline__ v16bf frag_b(const __bf16* t, int ld, int nbase, int k0){
  int lane = threadIdx.x & 31;
  const __bf16* p = t + (nbase + (lane&15))*ld + k0 + ((lane&16)?8:0);
  v16bf v;
  ((int4*)&v)[0] = *(const int4*)(p);
  ((int4*)&v)[1] = *(const int4*)(p+16);
  return v;
}

// ---------------------------------------------------------------------------
// Generic WMMA GEMM: Out[M,N] = act(A[M,K] @ B^T-packed[N,K] + bias)
// A rows mapped as: off(m) = (m>>rpgs)*gstride + (m&((1<<rpgs)-1))*rstride + abase
// Tile 128x128x32, 256 threads = 8 waves, wave tile 32x64 (2x4 WMMA accums).
// Double-buffered LDS, async global->LDS staging (4 async instr per wave/tile):
// issue tile t+1, s_wait_asynccnt 4 => tile t landed, barrier, WMMA on tile t.
// ---------------------------------------------------------------------------
template<int ACT, int OUTBF>
__global__ __launch_bounds__(256) void gemm_bf16_kernel(
    const __bf16* __restrict__ A, const __bf16* __restrict__ B,
    const float* __restrict__ bias, void* __restrict__ Out,
    int M, int N, int K,
    long long gstride, int rpgs, int rstride, long long abase)
{
  __shared__ alignas(16) __bf16 As[2][4096];   // 2 x 128x32
  __shared__ alignas(16) __bf16 Bs[2][4096];
  const int tid = threadIdx.x, lane = tid&31, w = tid>>5;
  const int wm = (w>>1)*32, wn = (w&1)*64;
  const long long m0 = (long long)blockIdx.y*128;
  const long long n0 = (long long)blockIdx.x*128;

  // K-invariant per-thread staging offsets (2 x 16B chunks for A and for B)
  long long aoff[2], boff[2];
  #pragma unroll
  for(int u=0;u<2;++u){
    int c=tid+u*256;
    int row=c>>2, kc=(c&3)*8;
    long long m=m0+row; if(m>=M)m=M-1;               // clamp (stores guarded)
    aoff[u]=(m>>rpgs)*gstride + (long long)(m&((1<<rpgs)-1))*rstride + abase + kc;
    long long n=n0+row; if(n>=N)n=N-1;
    boff[u]=n*(long long)K + kc;
  }
  const unsigned aLds = lds_addr(&As[0][0]) + tid*16;
  const unsigned bLds = lds_addr(&Bs[0][0]) + tid*16;

  const v8f vz = {0.f,0.f,0.f,0.f,0.f,0.f,0.f,0.f};
  v8f acc[2][4];
  for(int i=0;i<2;++i) for(int j=0;j<4;++j) acc[i][j]=vz;

  auto issue=[&](int k0,int buf){
    #pragma unroll
    for(int u=0;u<2;++u){
      async_b128(aLds + buf*8192 + u*4096, A + aoff[u] + k0);
      async_b128(bLds + buf*8192 + u*4096, B + boff[u] + k0);
    }
  };

  const int nt = K>>5;
  issue(0,0);
  for(int t=0;t<nt;++t){
    const int buf=t&1;
    if(t+1<nt){
      issue((t+1)<<5, buf^1);
      asm volatile("s_wait_asynccnt 0x4" ::: "memory");  // tile t complete
    }else{
      asm volatile("s_wait_asynccnt 0x0" ::: "memory");
    }
    __syncthreads();
    #pragma unroll
    for(int i=0;i<2;++i){
      v16bf af = frag_a(As[buf],32,wm+i*16,0);
      #pragma unroll
      for(int j=0;j<4;++j){
        v16bf bf = frag_b(Bs[buf],32,wn+j*16,0);
        acc[i][j]=WMMA_BF16(af,bf,acc[i][j]);
      }
    }
    __syncthreads();   // all waves done reading buf before it is re-filled
  }

  for(int i=0;i<2;++i) for(int j=0;j<4;++j){
    int col=(int)n0 + wn + j*16 + (lane&15);
    float bv=(bias && col<N)? bias[col] : 0.f;
    #pragma unroll
    for(int r=0;r<8;++r){
      int rowg=(int)m0 + wm + i*16 + r + ((lane>>4)<<3);
      if(rowg<M && col<N){
        float v=acc[i][j][r]+bv;
        if(ACT==1) v=siluf(v);
        if(OUTBF) ((__bf16*)Out)[(long long)rowg*N+col]=f2bf(v);
        else      ((float *)Out)[(long long)rowg*N+col]=v;
      }
    }
  }
}

// ------------------------- weight repack (fp32 [K][N] -> bf16 [N][K]) -------
__global__ void pack_wt_kernel(const float* __restrict__ W, __bf16* __restrict__ O,
                               int K, int N){
  long long i=(long long)blockIdx.x*blockDim.x+threadIdx.x;
  if(i>=(long long)K*N) return;
  long long n=i/K, k=i%K;
  O[i]=f2bf(W[k*(long long)N+n]);
}
__global__ void pack_tg_kernel(const float* __restrict__ tg, __bf16* __restrict__ o){
  int i=blockIdx.x*blockDim.x+threadIdx.x; if(i>=112*32)return;
  int r=i>>5,k=i&31;
  o[i]=f2bf((r<100&&k<25)? tg[r*25+k]:0.f);
}
__global__ void pack_fg_kernel(const float* __restrict__ fg, __bf16* __restrict__ o){
  int i=blockIdx.x*blockDim.x+threadIdx.x; if(i>=32*128)return;
  int r=i>>7,g=i&127;
  o[i]=f2bf((r<25&&g<100)? fg[g*25+r]:0.f);
}

// ------------------------- rms_norm_sh (+optional sys embed) ----------------
static __device__ __forceinline__ int l_of_coef(int i){
  if(i==0)return 0; if(i<4)return 1; if(i<9)return 2; if(i<16)return 3; return 4;
}
__global__ __launch_bounds__(128) void rms_norm_sh_kernel(
    const float* __restrict__ X, const float* __restrict__ Wn,
    const float* __restrict__ sys, __bf16* __restrict__ Out)
{
  int n=blockIdx.x, c=threadIdx.x;
  const float* xp = X + (long long)n*3200 + c;
  float s=0.f;
  #pragma unroll
  for(int i=0;i<25;++i){
    float v=xp[i*128]; int l=l_of_coef(i);
    s += v*v*(1.0f/((2*l+1)*5.0f));
  }
  __shared__ float red[128];
  red[c]=s; __syncthreads();
  for(int o=64;o>0;o>>=1){ if(c<o) red[c]+=red[c+o]; __syncthreads(); }
  float inv = rsqrtf(red[0]/128.0f + 1e-5f);
  #pragma unroll
  for(int i=0;i<25;++i){
    int l=l_of_coef(i);
    float v=xp[i*128]*inv*Wn[l*128+c];
    if(i==0 && sys) v += sys[(long long)n*128+c];
    Out[(long long)n*3200 + i*128 + c]=f2bf(v);
  }
}

// ------------------------- radial MLP layers 1-2 (64 ch, LN + silu) ---------
__global__ __launch_bounds__(64) void radial_mlp_kernel(
    const float* __restrict__ XE,
    const float* __restrict__ rw1,const float* __restrict__ rb1,
    const float* __restrict__ ls1,const float* __restrict__ lb1,
    const float* __restrict__ rw2,const float* __restrict__ rb2,
    const float* __restrict__ ls2,const float* __restrict__ lb2,
    __bf16* __restrict__ O)
{
  int e=blockIdx.x, c=threadIdx.x;
  __shared__ float xs[64], red[64], hs[64];
  xs[c]=XE[(long long)e*64+c];
  __syncthreads();
  float h=rb1[c];
  for(int k=0;k<64;++k) h += xs[k]*rw1[k*64+c];
  red[c]=h; __syncthreads();
  for(int o=32;o>0;o>>=1){ if(c<o) red[c]+=red[c+o]; __syncthreads(); }
  float mu=red[0]/64.f; __syncthreads();
  red[c]=(h-mu)*(h-mu); __syncthreads();
  for(int o=32;o>0;o>>=1){ if(c<o) red[c]+=red[c+o]; __syncthreads(); }
  float var=red[0]/64.f; __syncthreads();
  hs[c]=siluf((h-mu)*rsqrtf(var+1e-5f)*ls1[c]+lb1[c]);
  __syncthreads();
  float h2=rb2[c];
  for(int k=0;k<64;++k) h2 += hs[k]*rw2[k*64+c];
  red[c]=h2; __syncthreads();
  for(int o=32;o>0;o>>=1){ if(c<o) red[c]+=red[c+o]; __syncthreads(); }
  mu=red[0]/64.f; __syncthreads();
  red[c]=(h2-mu)*(h2-mu); __syncthreads();
  for(int o=32;o>0;o>>=1){ if(c<o) red[c]+=red[c+o]; __syncthreads(); }
  var=red[0]/64.f;
  O[(long long)e*64+c]=f2bf(siluf((h2-mu)*rsqrtf(var+1e-5f)*ls2[c]+lb2[c]));
}

// rad slice base per m-row (channels of 256), broadcast over real/imag rows
static __device__ __forceinline__ int rad_base(int r){
  if(r<5)  return r*256;
  if(r<13) return 1280 + ((r-5)&3)*256;
  int q=r-13; if(q>=3)q-=3; return 2304 + q*256;
}

// ------------------------- per-edge Wigner rotate (WMMA) --------------------
// xm[e] (19x256) = W[e](19x25) @ concat(h[src],h[tgt]) (25x256), then *rad, bf16
__global__ __launch_bounds__(256) void wigner_fwd_kernel(
    const float* __restrict__ WIG, const __bf16* __restrict__ H,
    const int* __restrict__ SRC, const int* __restrict__ TGT,
    const __bf16* __restrict__ RAD, __bf16* __restrict__ XM)
{
  int e=blockIdx.x, tid=threadIdx.x, lane=tid&31, w=tid>>5;
  __shared__ alignas(16) __bf16 Aw[32*32];
  __shared__ alignas(16) __bf16 Bh[256*32];
  __shared__ alignas(16) __bf16 RadS[3072];
  const float* wp = WIG + (long long)e*475;
  for(int i=tid;i<1024;i+=256){
    int r=i>>5,k=i&31;
    Aw[i]=f2bf((r<19&&k<25)? wp[r*25+k]:0.f);
  }
  // per-edge rad vector: 384 x b128 into LDS (read back in epilogue)
  {
    const int4* rp=(const int4*)(RAD + (long long)e*3072);
    ((int4*)RadS)[tid] = rp[tid];
    if(tid<128) ((int4*)RadS)[tid+256] = rp[tid+256];
  }
  int s=SRC[e], t=TGT[e];
  // b128 loads of h rows (8 bf16), transposed on LDS store side
  for(int c=tid;c<800;c+=256){
    int half=(c>=400)?1:0, cc=c-half*400;
    int k=cc>>4, chx=(cc&15)*8;
    int node = half? t : s;
    int4 q = *(const int4*)(H + (long long)node*3200 + k*128 + chx);
    const __bf16* qb=(const __bf16*)&q;
    #pragma unroll
    for(int j=0;j<8;++j) Bh[(half*128+chx+j)*32 + k] = qb[j];
  }
  { int ch=tid; for(int k=25;k<32;++k) Bh[ch*32+k]=(__bf16)0.f; }  // K pad
  __syncthreads();
  const v8f vz={0.f,0.f,0.f,0.f,0.f,0.f,0.f,0.f};
  v8f acc[2][2]; for(int i=0;i<2;++i)for(int j=0;j<2;++j)acc[i][j]=vz;
  #pragma unroll
  for(int i=0;i<2;++i){
    v16bf af=frag_a(Aw,32,i*16,0);
    #pragma unroll
    for(int j=0;j<2;++j){
      v16bf bf=frag_b(Bh,32,w*32+j*16,0);
      acc[i][j]=WMMA_BF16(af,bf,acc[i][j]);
    }
  }
  for(int i=0;i<2;++i)for(int j=0;j<2;++j){
    int col=w*32+j*16+(lane&15);
    #pragma unroll
    for(int r=0;r<8;++r){
      int row=i*16+r+((lane>>4)<<3);
      if(row<19){
        float rv=bf2f(RadS[rad_base(row) + col]);
        XM[(long long)e*4864 + row*256 + col]=f2bf(acc[i][j][r]*rv);
      }
    }
  }
}

// ------------------------- gate_act (conv1 outputs -> conv2 input) ----------
__global__ __launch_bounds__(128) void gate_kernel(
    const float* __restrict__ Y0, const float* __restrict__ Y1,
    const float* __restrict__ Y2, __bf16* __restrict__ X2)
{
  int r=blockIdx.x, e=blockIdx.y, c=threadIdx.x;
  long long e0=(long long)e*1152, e1=(long long)(2*e)*1024, e2=(long long)(2*e)*768;
  float val; int gl=-1;
  if(r==0){ val=siluf(Y0[e0+512+c]); }
  else if(r<5){ val=Y0[e0+512+r*128+c]; gl=r-1; }
  else if(r<9){ int q=r-5;  val=Y1[e1+q*128+c]       - Y1[e1+1024+512+q*128+c]; gl=q; }
  else if(r<13){int q=r-9;  val=Y1[e1+1024+q*128+c]  + Y1[e1+512+q*128+c];      gl=q; }
  else if(r<16){int q=r-13; val=Y2[e2+q*128+c]       - Y2[e2+768+384+q*128+c];  gl=q+1; }
  else        { int q=r-16; val=Y2[e2+768+q*128+c]   + Y2[e2+384+q*128+c];      gl=q+1; }
  if(gl>=0) val *= sigmf(Y0[e0+gl*128+c]);
  X2[(long long)e*2432 + r*128 + c]=f2bf(val);
}

// -------- conv2 recombine * envelope -> y2 [e][19][128] bf16 (coalesced) ----
__global__ __launch_bounds__(128) void recomb_kernel(
    const float* __restrict__ Z0, const float* __restrict__ Z1,
    const float* __restrict__ Z2, const float* __restrict__ ENV,
    __bf16* __restrict__ Y2)
{
  int r=blockIdx.x, e=blockIdx.y, c=threadIdx.x;
  long long e0=(long long)e*640, e1=(long long)(2*e)*1024, e2=(long long)(2*e)*768;
  float val;
  if(r<5)       val=Z0[e0+r*128+c];
  else if(r<9){ int q=r-5;  val=Z1[e1+q*128+c]      - Z1[e1+1024+512+q*128+c]; }
  else if(r<13){int q=r-9;  val=Z1[e1+1024+q*128+c] + Z1[e1+512+q*128+c]; }
  else if(r<16){int q=r-13; val=Z2[e2+q*128+c]      - Z2[e2+768+384+q*128+c]; }
  else        { int q=r-16; val=Z2[e2+768+q*128+c]  + Z2[e2+384+q*128+c]; }
  val *= ENV[e];
  Y2[(long long)e*2432 + r*128 + c]=f2bf(val);
}

// ------------------------- per-edge inverse Wigner + segment-sum ------------
__global__ __launch_bounds__(256) void wigner_inv_kernel(
    const float* __restrict__ WINV, const __bf16* __restrict__ Y2,
    const int* __restrict__ TGT, float* __restrict__ XA)
{
  int e=blockIdx.x, tid=threadIdx.x, lane=tid&31, w=tid>>5;
  __shared__ alignas(16) __bf16 Aw[32*32];
  __shared__ alignas(16) __bf16 Bt[128*32];
  const float* wp = WINV + (long long)e*475;
  for(int i=tid;i<1024;i+=256){
    int r=i>>5,k=i&31;
    Aw[i]=f2bf((r<25&&k<19)? wp[r*19+k]:0.f);
  }
  // y2[e] rows (19x128) -> Bt[ch][row], b128 loads + transpose on LDS store
  for(int c=tid;c<304;c+=256){
    int row=c>>4, chx=(c&15)*8;
    int4 q=*(const int4*)(Y2 + (long long)e*2432 + row*128 + chx);
    const __bf16* qb=(const __bf16*)&q;
    #pragma unroll
    for(int j=0;j<8;++j) Bt[(chx+j)*32 + row] = qb[j];
  }
  { int ch=tid; if(ch<128){ for(int k=19;k<32;++k) Bt[ch*32+k]=(__bf16)0.f; } }
  __syncthreads();
  const v8f vz={0.f,0.f,0.f,0.f,0.f,0.f,0.f,0.f};
  v8f acc[2]; acc[0]=vz; acc[1]=vz;
  #pragma unroll
  for(int i=0;i<2;++i){
    v16bf af=frag_a(Aw,32,i*16,0);
    v16bf bf=frag_b(Bt,32,w*16,0);
    acc[i]=WMMA_BF16(af,bf,acc[i]);
  }
  int t=TGT[e];
  for(int i=0;i<2;++i){
    int col=w*16+(lane&15);
    #pragma unroll
    for(int r=0;r<8;++r){
      int row=i*16+r+((lane>>4)<<3);
      if(row<25) atomicAdd(&XA[(long long)t*3200 + row*128 + col], acc[i][r]);
    }
  }
}

// ------------------------- grid FFN: to_grid / from_grid (WMMA) -------------
__global__ __launch_bounds__(256) void togrid_kernel(
    const __bf16* __restrict__ TGP, const __bf16* __restrict__ H2,
    __bf16* __restrict__ XG)
{
  int n=blockIdx.x, tid=threadIdx.x, lane=tid&31, w=tid>>5;
  __shared__ alignas(16) __bf16 At[112*32];
  __shared__ alignas(16) __bf16 Bt[128*32];
  for(int c=tid;c<448;c+=256) ((int4*)At)[c]=((const int4*)TGP)[c];
  for(int c=tid;c<400;c+=256){
    int k=c>>4, chx=(c&15)*8;
    int4 q=*(const int4*)(H2 + (long long)n*3200 + k*128 + chx);
    const __bf16* qb=(const __bf16*)&q;
    #pragma unroll
    for(int j=0;j<8;++j) Bt[(chx+j)*32+k]=qb[j];
  }
  { int ch=tid; if(ch<128){ for(int k=25;k<32;++k) Bt[ch*32+k]=(__bf16)0.f; } }
  __syncthreads();
  const v8f vz={0.f,0.f,0.f,0.f,0.f,0.f,0.f,0.f};
  v8f acc[7]; for(int i=0;i<7;++i)acc[i]=vz;
  v16bf bf=frag_b(Bt,32,w*16,0);
  #pragma unroll
  for(int i=0;i<7;++i){
    v16bf af=frag_a(At,32,i*16,0);
    acc[i]=WMMA_BF16(af,bf,acc[i]);
  }
  for(int i=0;i<7;++i){
    int col=w*16+(lane&15);
    #pragma unroll
    for(int r=0;r<8;++r){
      int g=i*16+r+((lane>>4)<<3);
      if(g<100) XG[(long long)n*12800 + g*128 + col]=f2bf(acc[i][r]);
    }
  }
}

__global__ __launch_bounds__(256) void fromgrid_kernel(
    const __bf16* __restrict__ FGP, const __bf16* __restrict__ G3,
    const float* __restrict__ XA, float* __restrict__ OUT)
{
  int n=blockIdx.x, tid=threadIdx.x, lane=tid&31, w=tid>>5;
  __shared__ alignas(16) __bf16 Af[32*128];
  __shared__ alignas(16) __bf16 Bt[128*128];
  for(int c=tid;c<512;c+=256) ((int4*)Af)[c]=((const int4*)FGP)[c];
  for(int c=tid;c<1600;c+=256){
    int g=c>>4, chx=(c&15)*8;
    int4 q=*(const int4*)(G3 + (long long)n*12800 + g*128 + chx);
    const __bf16* qb=(const __bf16*)&q;
    #pragma unroll
    for(int j=0;j<8;++j) Bt[(chx+j)*128+g]=qb[j];
  }
  for(int i=tid;i<4096;i+=256){       // pad g = 100..127
    int ch=i>>5, g=100+(i&31);
    if(g<128) Bt[ch*128+g]=(__bf16)0.f;
  }
  __syncthreads();
  const v8f vz={0.f,0.f,0.f,0.f,0.f,0.f,0.f,0.f};
  v8f acc[2]; acc[0]=vz; acc[1]=vz;
  #pragma unroll
  for(int k0=0;k0<128;k0+=32){
    #pragma unroll
    for(int i=0;i<2;++i){
      v16bf af=frag_a(Af,128,i*16,k0);
      v16bf bf=frag_b(Bt,128,w*16,k0);
      acc[i]=WMMA_BF16(af,bf,acc[i]);
    }
  }
  for(int i=0;i<2;++i){
    int col=w*16+(lane&15);
    #pragma unroll
    for(int r=0;r<8;++r){
      int row=i*16+r+((lane>>4)<<3);
      if(row<25){
        long long idx=(long long)n*3200 + row*128 + col;
        OUT[idx]=acc[i][r]+XA[idx];
      }
    }
  }
}

// ---------------------------------------------------------------------------
static inline void launch_gemm(int ACT,int OUTBF,
    const __bf16*A,const __bf16*B,const float*bias,void*O,
    int M,int N,int K,long long gs,int rpgs,int rs,long long ab,hipStream_t st)
{
  dim3 g(N/128,(M+127)/128);
  if(ACT==0&&OUTBF==0)      gemm_bf16_kernel<0,0><<<g,256,0,st>>>(A,B,bias,O,M,N,K,gs,rpgs,rs,ab);
  else if(ACT==0&&OUTBF==1) gemm_bf16_kernel<0,1><<<g,256,0,st>>>(A,B,bias,O,M,N,K,gs,rpgs,rs,ab);
  else                      gemm_bf16_kernel<1,1><<<g,256,0,st>>>(A,B,bias,O,M,N,K,gs,rpgs,rs,ab);
}

extern "C" void kernel_launch(void* const* d_in, const int* in_sizes, int n_in,
                              void* d_out, int out_size, void* d_ws, size_t ws_size,
                              hipStream_t stream)
{
  (void)in_sizes; (void)n_in; (void)out_size; (void)ws_size;
  // setup_inputs() dict order, params flattened depth-first in insertion order
  const float* X    =(const float*)d_in[0];
  const float* XE   =(const float*)d_in[1];
  /* d_in[2] edge_distance: unused by reference */
  const int*   EI   =(const int*)  d_in[3];
  const float* WIG  =(const float*)d_in[4];
  const float* WINV =(const float*)d_in[5];
  const float* ENV  =(const float*)d_in[6];
  const float* SYS  =(const float*)d_in[7];
  const float* N1W  =(const float*)d_in[8];
  const float* N2W  =(const float*)d_in[9];
  const float* RW1=(const float*)d_in[10]; const float* RB1=(const float*)d_in[11];
  const float* LS1=(const float*)d_in[12]; const float* LB1=(const float*)d_in[13];
  const float* RW2=(const float*)d_in[14]; const float* RB2=(const float*)d_in[15];
  const float* LS2=(const float*)d_in[16]; const float* LB2=(const float*)d_in[17];
  const float* RW3=(const float*)d_in[18]; const float* RB3=(const float*)d_in[19];
  const float* W1M0=(const float*)d_in[20]; const float* B1M0=(const float*)d_in[21];
  const float* W1M1=(const float*)d_in[22]; const float* W1M2=(const float*)d_in[23];
  const float* W2M0=(const float*)d_in[24]; const float* B2M0=(const float*)d_in[25];
  const float* W2M1=(const float*)d_in[26]; const float* W2M2=(const float*)d_in[27];
  const float* TG =(const float*)d_in[28]; const float* FG =(const float*)d_in[29];
  const float* FW1=(const float*)d_in[30]; const float* FW2=(const float*)d_in[31];
  const float* FW3=(const float*)d_in[32];
  const int* SRC=EI; const int* TGT=EI+16000;

  // ---------------- workspace layout (regions reused across phases) --------
  size_t cur=0;
  auto take=[&](size_t bytes)->void*{
    void* p=(char*)d_ws+cur; cur=(cur+bytes+255)&~(size_t)255; return p;
  };
  __bf16* w1m0p=(__bf16*)take(1152LL*1280*2);
  __bf16* w1m1p=(__bf16*)take(1024LL*1024*2);
  __bf16* w1m2p=(__bf16*)take(768LL*768*2);
  __bf16* w2m0p=(__bf16*)take(640LL*640*2);
  __bf16* w2m1p=(__bf16*)take(1024LL*512*2);
  __bf16* w2m2p=(__bf16*)take(768LL*384*2);
  __bf16* wr3p =(__bf16*)take(3072LL*64*2);
  __bf16* wf1p =(__bf16*)take(128LL*128*2);
  __bf16* wf2p =(__bf16*)take(128LL*128*2);
  __bf16* wf3p =(__bf16*)take(128LL*128*2);
  __bf16* tgp  =(__bf16*)take(112LL*32*2);
  __bf16* fgp  =(__bf16*)take(32LL*128*2);
  __bf16* h1   =(__bf16*)take(2000LL*3200*2);    // normed nodes (bf16)
  __bf16* h2rad=(__bf16*)take(16000LL*64*2);     // radial hidden
  __bf16* x2   =(__bf16*)take(16000LL*2432*2);   // gated conv2 input
  float*  xa   =(float*) take(2000LL*3200*4);    // node accumulator (residual+msgs)
  __bf16* h2n  =(__bf16*)take(2000LL*3200*2);    // normed nodes #2
  float*  rA   =(float*) take(16000LL*1152*4);   // ym0 / zm0
  float*  rB   =(float*) take(32000LL*1024*4);   // ym1 / zm1
  float*  rC   =(float*) take(32000LL*768*4);    // ym2 / zm2
  char*   rD   =(char*)  take(16000LL*4864*2);   // xm1 -> y2 -> g1,g2,g3
  char*   rE   =(char*)  take(16000LL*3072*2);   // rad -> xg
  __bf16* xm1=(__bf16*)rD;
  __bf16* y2 =(__bf16*)rD;                       // reuses xm1 (dead after conv1 GEMMs)
  __bf16* g1 =(__bf16*)rD;                       // reuses y2 (dead after wigner_inv)
  __bf16* g2 =(__bf16*)(rD+ 51200000);
  __bf16* g3 =(__bf16*)(rD+102400000);
  __bf16* rad=(__bf16*)rE;
  __bf16* xg =(__bf16*)rE;                       // reuses rad (dead after wigner_fwd)

  // ---------------- weight repack (fp32 [K][N] -> bf16 [N][K]) -------------
  auto packw=[&](const float*W,__bf16*O,int K,int N){
    long long tot=(long long)K*N;
    pack_wt_kernel<<<(unsigned)((tot+255)/256),256,0,stream>>>(W,O,K,N);
  };
  packw(W1M0,w1m0p,1280,1152); packw(W1M1,w1m1p,1024,1024); packw(W1M2,w1m2p,768,768);
  packw(W2M0,w2m0p,640,640);   packw(W2M1,w2m1p,512,1024);  packw(W2M2,w2m2p,384,768);
  packw(RW3,wr3p,64,3072);
  packw(FW1,wf1p,128,128); packw(FW2,wf2p,128,128); packw(FW3,wf3p,128,128);
  pack_tg_kernel<<<14,256,0,stream>>>(TG,tgp);
  pack_fg_kernel<<<16,256,0,stream>>>(FG,fgp);

  // ---------------- pipeline ------------------------------------------------
  rms_norm_sh_kernel<<<2000,128,0,stream>>>(X,N1W,SYS,h1);
  radial_mlp_kernel<<<16000,64,0,stream>>>(XE,RW1,RB1,LS1,LB1,RW2,RB2,LS2,LB2,h2rad);
  // rad = h2rad @ rw3 + rb3 (M=16000,K=64,N=3072), bf16 out
  launch_gemm(0,1,h2rad,wr3p,RB3,rad,16000,3072,64, 64,0,0,0, stream);
  wigner_fwd_kernel<<<16000,256,0,stream>>>(WIG,h1,SRC,TGT,rad,xm1);
  // conv1 GEMMs (A = xm1 [e][19][256])
  launch_gemm(0,0,xm1,w1m0p,B1M0,rA,16000,1152,1280, 4864,0,0,   0,   stream);
  launch_gemm(0,0,xm1,w1m1p,0,   rB,32000,1024,1024, 4864,1,1024,1280,stream);
  launch_gemm(0,0,xm1,w1m2p,0,   rC,32000, 768, 768, 4864,1, 768,3328,stream);
  gate_kernel<<<dim3(19,16000),128,0,stream>>>(rA,rB,rC,x2);
  // conv2 GEMMs (A = x2 [e][19][128])
  launch_gemm(0,0,x2,w2m0p,B2M0,rA,16000, 640,640, 2432,0,0,  0,   stream);
  launch_gemm(0,0,x2,w2m1p,0,   rB,32000,1024,512, 2432,1,512,640, stream);
  launch_gemm(0,0,x2,w2m2p,0,   rC,32000, 768,384, 2432,1,384,1664,stream);
  recomb_kernel<<<dim3(19,16000),128,0,stream>>>(rA,rB,rC,ENV,y2);
  hipMemcpyAsync(xa,X,2000LL*3200*4,hipMemcpyDeviceToDevice,stream); // residual
  wigner_inv_kernel<<<16000,256,0,stream>>>(WINV,y2,TGT,xa);         // segment-sum
  rms_norm_sh_kernel<<<2000,128,0,stream>>>(xa,N2W,nullptr,h2n);
  togrid_kernel<<<2000,256,0,stream>>>(tgp,h2n,xg);
  launch_gemm(1,1,xg,wf1p,0,g1,200000,128,128, 128,0,0,0, stream);  // silu
  launch_gemm(1,1,g1,wf2p,0,g2,200000,128,128, 128,0,0,0, stream);  // silu
  launch_gemm(0,1,g2,wf3p,0,g3,200000,128,128, 128,0,0,0, stream);
  fromgrid_kernel<<<2000,256,0,stream>>>(fgp,g3,xa,(float*)d_out);
}
